// FasterRCNN_46462956208215
// MI455X (gfx1250) — matrix operand order repaired and checked by reading, not compiled, not run on hardware
//
#include <hip/hip_runtime.h>
#include <cmath>

// ---------------------------------------------------------------------------
// CDNA5 WMMA types (wave32): A/B v16h (8 VGPRs), C/D v8f (8 VGPRs)
// ---------------------------------------------------------------------------
typedef _Float16 h16;
typedef __attribute__((ext_vector_type(16))) _Float16 v16h;
typedef __attribute__((ext_vector_type(8)))  _Float16 v8h;
typedef __attribute__((ext_vector_type(8)))  float    v8f;
typedef __attribute__((ext_vector_type(4)))  float    f4;

union U16h { v16h v; v8h h[2]; };

#define DEVI __device__ __forceinline__

DEVI int imaxi(int a, int b) { return a > b ? a : b; }
DEVI int imini(int a, int b) { return a < b ? a : b; }

// 16-bit A-matrix 16x32 per-lane K map: K = 16g + i, lane group g = lane>>4
DEVI int kmapA(int i, int g) {
    return i + (g << 4);
}
// 16-bit B-matrix 32x16: K = 16g + i
DEVI int kmapB(int i, int g) { return i + (g << 4); }

// ---------------------------------------------------------------------------
// CDNA5 async global->LDS copy (ASYNCcnt domain), ISA §10.7/§15.18.3.
// VDST VGPR = raw LDS byte offset. Generic shared pointers map to LDS by
// their low 32 bits.
// ---------------------------------------------------------------------------
DEVI void async_load_b128_to_lds(void* lds_ptr, const void* gptr) {
    unsigned lds_off = (unsigned)(uintptr_t)lds_ptr;
    asm volatile("global_load_async_to_lds_b128 %0, %1, off"
                 :: "v"(lds_off), "v"((unsigned long long)(uintptr_t)gptr)
                 : "memory");
}
DEVI void wait_async0() {
    asm volatile("s_wait_asynccnt 0x0" ::: "memory");
}

// ---------------------------------------------------------------------------
// Packing
// ---------------------------------------------------------------------------
__global__ void k_pack_input(const float* __restrict__ im, h16* __restrict__ out, int n) {
    int t = blockIdx.x * blockDim.x + threadIdx.x;
    if (t < n) out[t] = (h16)im[t];
}

// OIHW f32 -> [o][k = tap*Cin + ci] f16  (tap = ky*3+kx)
__global__ void k_pack_w3x3(const float* __restrict__ w, h16* __restrict__ wp,
                            int Cin, int Cout) {
    int t = blockIdx.x * blockDim.x + threadIdx.x;
    int K = Cin * 9;
    if (t >= K * Cout) return;
    int k = t % K, o = t / K;
    int tap = k / Cin, ci = k - tap * Cin;
    int ky = tap / 3, kx = tap - ky * 3;
    wp[(size_t)o * K + k] =
        (h16)w[(((size_t)o * Cin + ci) * 3 + (size_t)ky) * 3 + kx];
}

// ---------------------------------------------------------------------------
// Block-cooperative 3x3 SAME conv on WMMA with async-LDS-staged weights.
// Cin % 32 == 0, Cout % 64 == 0. Block = 8 waves sharing one 64-ch output
// group; wave w -> pixel tile (blockIdx.x*8 + w). Weight chunk (64ch x 32k,
// 4 KB f16) double-buffered in LDS via global_load_async_to_lds_b128 (one per
// thread per chunk, ASYNCcnt-tracked), consumed with ds_load_b128.
// ---------------------------------------------------------------------------
__global__ __launch_bounds__(256)
void k_conv3x3_wmma_lds(const h16* __restrict__ in, const h16* __restrict__ wp,
                        const float* __restrict__ bias, h16* __restrict__ out,
                        int Cin, int H, int W, int Cout, int relu) {
    __shared__ h16 ldsB[2][64 * 32];
    const int K = Cin * 9;
    const int HW = H * W;
    const int tid = threadIdx.x;
    const int waveid = tid >> 5;
    const int lane = tid & 31;
    const int g = lane >> 4;
    const int l15 = lane & 15;
    const int mt = blockIdx.x * 8 + waveid;
    const int ngBase = blockIdx.y * 64;
    const int m = mt * 16 + l15;                 // output pixel (A row)
    const int mm = (m < HW) ? m : (HW - 1);
    const int y = mm / W, x = mm % W;

    // weight-fill assignment: thread -> (channel, 8-half slice)
    const int fch = tid >> 2;                    // 0..63
    const int fpart = (tid & 3) << 3;            // 0,8,16,24
    const h16* wrow = wp + (size_t)(ngBase + fch) * K + fpart;
    h16* ldsDst = &ldsB[0][fch * 32 + fpart];    // buffer 0 slot (buffer 1 = +2048)

    const int nchunk = (Cin >> 5) * 9;

    // fill chunk 0 (tap 0, cb 0) asynchronously
    async_load_b128_to_lds(ldsDst, wrow);
    wait_async0();
    __syncthreads();

    v8f acc0 = {}, acc1 = {}, acc2 = {}, acc3 = {};
    const v8h z8 = {};

    int tap = 0, cb = 0;                         // chunk being computed
    int ftap = 0, fcb = 32;                      // chunk being filled (c+1)
    if (fcb == Cin) { fcb = 0; ftap = 1; }

    for (int c = 0; c < nchunk; ++c) {
        // ---- async fill of chunk c+1 into buffer (c+1)&1, overlaps compute ----
        if (c + 1 < nchunk) {
            async_load_b128_to_lds(ldsDst + (((c + 1) & 1) << 11),
                                   wrow + ftap * Cin + fcb);
            fcb += 32;
            if (fcb == Cin) { fcb = 0; ++ftap; }
        }
        // ---- compute chunk c ----
        const int ky = tap / 3, kx = tap - (tap / 3) * 3;
        const int iy = y + ky - 1, ix = x + kx - 1;
        const bool ok = (m < HW) &&
                        ((unsigned)iy < (unsigned)H) && ((unsigned)ix < (unsigned)W);
        U16h a;
        a.h[0] = z8;
        a.h[1] = z8;
        if (ok) {                                // predicated b128 pair
            const h16* pa = in + (size_t)(iy * W + ix) * Cin + cb + (g << 4);
            a.h[0] = *(const v8h*)pa;
            a.h[1] = *(const v8h*)(pa + 8);
        }
        const h16* lb = &ldsB[c & 1][g << 4];
        U16h b0, b1, b2, b3;
        b0.h[0] = *(const v8h*)(lb + (l15 +  0) * 32);
        b0.h[1] = *(const v8h*)(lb + (l15 +  0) * 32 + 8);
        b1.h[0] = *(const v8h*)(lb + (l15 + 16) * 32);
        b1.h[1] = *(const v8h*)(lb + (l15 + 16) * 32 + 8);
        b2.h[0] = *(const v8h*)(lb + (l15 + 32) * 32);
        b2.h[1] = *(const v8h*)(lb + (l15 + 32) * 32 + 8);
        b3.h[0] = *(const v8h*)(lb + (l15 + 48) * 32);
        b3.h[1] = *(const v8h*)(lb + (l15 + 48) * 32 + 8);
        acc0 = __builtin_amdgcn_wmma_f32_16x16x32_f16(false, a.v, false, b0.v, (short)0, acc0, false, false);
        acc1 = __builtin_amdgcn_wmma_f32_16x16x32_f16(false, a.v, false, b1.v, (short)0, acc1, false, false);
        acc2 = __builtin_amdgcn_wmma_f32_16x16x32_f16(false, a.v, false, b2.v, (short)0, acc2, false, false);
        acc3 = __builtin_amdgcn_wmma_f32_16x16x32_f16(false, a.v, false, b3.v, (short)0, acc3, false, false);
        cb += 32;
        if (cb == Cin) { cb = 0; ++tap; }
        wait_async0();                           // this wave's fill(c+1) complete
        __syncthreads();                         // all fills visible; compute(c) done
    }

    // D: VGPR r -> M = r + 8g, N = lane&15 ; output HWC
    const int om0 = mt * 16 + (g << 3);
    v8f accs[4] = {acc0, acc1, acc2, acc3};
#pragma unroll
    for (int t = 0; t < 4; ++t) {
        const int n = ngBase + t * 16 + l15;
        const float bv = bias[n];
#pragma unroll
        for (int r = 0; r < 8; ++r) {
            int om = om0 + r;
            if (om < HW) {
                float v = accs[t][r] + bv;
                if (relu) v = fmaxf(v, 0.0f);
                out[(size_t)om * Cout + n] = (h16)v;
            }
        }
    }
}

// ---------------------------------------------------------------------------
// Generic 3x3 conv for Cin == 3 (conv1_1 only, ~0.5% of FLOPs)
// ---------------------------------------------------------------------------
__global__ void k_conv3x3_any(const h16* __restrict__ in, const h16* __restrict__ wp,
                              const float* __restrict__ bias, h16* __restrict__ out,
                              int Cin, int H, int W, int Cout, int relu) {
    const int K = Cin * 9;
    const int HW = H * W;
    const int Mtiles = (HW + 15) >> 4;
    const int Ntiles = Cout >> 4;
    const int wave = blockIdx.x * (blockDim.x >> 5) + (threadIdx.x >> 5);
    if (wave >= Mtiles * Ntiles) return;
    const int lane = threadIdx.x & 31;
    const int mt = wave % Mtiles;
    const int nt = wave / Mtiles;
    const int g = lane >> 4;
    const int l15 = lane & 15;
    const int m = mt * 16 + l15;
    const int n = nt * 16 + l15;
    const int mm = (m < HW) ? m : HW - 1;
    const int y = mm / W, x = mm % W;
    const h16* wrow = wp + (size_t)n * K;

    v8f c = {};
    for (int kb = 0; kb < K; kb += 32) {
        v16h a, b;
#pragma unroll
        for (int i = 0; i < 16; ++i) {
            int ka = kb + kmapA(i, g);
            float av = 0.0f;
            if (ka < K && m < HW) {
                int tap = ka / Cin, ci = ka - tap * Cin;
                int ky = tap / 3, kx = tap - ky * 3;
                int iy = y + ky - 1, ix = x + kx - 1;
                if ((unsigned)iy < (unsigned)H && (unsigned)ix < (unsigned)W)
                    av = (float)in[(size_t)(iy * W + ix) * Cin + ci];
            }
            a[i] = (h16)av;
            int kbb = kb + kmapB(i, g);
            b[i] = (kbb < K) ? wrow[kbb] : (h16)0.0f;
        }
        c = __builtin_amdgcn_wmma_f32_16x16x32_f16(false, a, false, b, (short)0, c, false, false);
    }
    const int om0 = mt * 16 + (g << 3);
    const float bv = bias[n];
#pragma unroll
    for (int r = 0; r < 8; ++r) {
        int om = om0 + r;
        if (om < HW) {
            float v = c[r] + bv;
            if (relu) v = fmaxf(v, 0.0f);
            out[(size_t)om * Cout + n] = (h16)v;
        }
    }
}

// ---------------------------------------------------------------------------
// 2x2 stride-2 VALID max pool, HWC f16
// ---------------------------------------------------------------------------
__global__ void k_maxpool(const h16* __restrict__ in, h16* __restrict__ out,
                          int C, int H, int W) {
    int Ho = H >> 1, Wo = W >> 1;
    size_t total = (size_t)C * Ho * Wo;
    size_t t = (size_t)blockIdx.x * blockDim.x + threadIdx.x;
    if (t >= total) return;
    int c  = (int)(t % C);
    int wo = (int)((t / C) % Wo);
    int ho = (int)(t / ((size_t)C * Wo));
    const h16* p = in + ((size_t)(2 * ho) * W + 2 * wo) * C + c;
    const size_t rs = (size_t)W * C;
    float v = fmaxf(fmaxf((float)p[0], (float)p[C]),
                    fmaxf((float)p[rs], (float)p[rs + C]));
    out[t] = (h16)v;
}

// 1x1 conv heads (tiny N): scalar, CHW f32 output; HWC f16 input
__global__ void k_conv1x1(const h16* __restrict__ in, const float* __restrict__ w,
                          const float* __restrict__ b, float* __restrict__ out,
                          int Cin, int HW, int Cout) {
    int t = blockIdx.x * blockDim.x + threadIdx.x;
    if (t >= Cout * HW) return;
    int p = t % HW, co = t / HW;
    const float* wr = w + (size_t)co * Cin;
    const h16* ir = in + (size_t)p * Cin;
    float s = b[co];
    for (int ci = 0; ci < Cin; ++ci) s += wr[ci] * (float)ir[ci];
    out[t] = s;
}

// ---------------------------------------------------------------------------
// Proposal decode (idx = (y*W+x)*9 + a)
// ---------------------------------------------------------------------------
struct Anchors { float a[36]; };

__global__ void k_decode(const float* __restrict__ score, const float* __restrict__ bbox,
                         const float* __restrict__ im_info, Anchors AN,
                         float* __restrict__ scores, float* __restrict__ boxes,
                         int H, int W) {
    int HW = H * W;
    int t = blockIdx.x * blockDim.x + threadIdx.x;
    if (t >= HW * 9) return;
    int a = t % 9, p = t / 9;
    int py = p / W, px = p % W;
    float sx = px * 16.0f, sy = py * 16.0f;
    float ax1 = AN.a[a * 4 + 0] + sx, ay1 = AN.a[a * 4 + 1] + sy;
    float ax2 = AN.a[a * 4 + 2] + sx, ay2 = AN.a[a * 4 + 3] + sy;
    float w = ax2 - ax1 + 1.0f, h = ay2 - ay1 + 1.0f;
    float cx = ax1 + 0.5f * w, cy = ay1 + 0.5f * h;
    float d0 = bbox[(size_t)(a * 4 + 0) * HW + p];
    float d1 = bbox[(size_t)(a * 4 + 1) * HW + p];
    float d2 = bbox[(size_t)(a * 4 + 2) * HW + p];
    float d3 = bbox[(size_t)(a * 4 + 3) * HW + p];
    float pcx = d0 * w + cx, pcy = d1 * h + cy;
    float pw = expf(d2) * w, ph = expf(d3) * h;
    float imH = im_info[0], imW = im_info[1], imS = im_info[2];
    float x1 = fminf(fmaxf(pcx - 0.5f * pw, 0.0f), imW - 1.0f);
    float y1 = fminf(fmaxf(pcy - 0.5f * ph, 0.0f), imH - 1.0f);
    float x2 = fminf(fmaxf(pcx + 0.5f * pw, 0.0f), imW - 1.0f);
    float y2 = fminf(fmaxf(pcy + 0.5f * ph, 0.0f), imH - 1.0f);
    boxes[(size_t)t * 4 + 0] = x1; boxes[(size_t)t * 4 + 1] = y1;
    boxes[(size_t)t * 4 + 2] = x2; boxes[(size_t)t * 4 + 3] = y2;
    float sb = score[(size_t)a * HW + p];
    float sf = score[(size_t)(9 + a) * HW + p];
    float sc = 1.0f / (1.0f + expf(-(sf - sb)));
    float bw = x2 - x1 + 1.0f, bh = y2 - y1 + 1.0f;
    float ms = 16.0f * imS;
    scores[t] = (bw >= ms && bh >= ms) ? sc : -__builtin_inff();
}

// Exact stable top-k via rank (score desc, index asc)
__global__ void k_rank(const float* __restrict__ s, int n, int* __restrict__ rank) {
    int i = blockIdx.x * blockDim.x + threadIdx.x;
    if (i >= n) return;
    float si = s[i];
    int r = 0;
    for (int j = 0; j < n; ++j) {
        float sj = s[j];
        r += (sj > si) || (sj == si && j < i);
    }
    rank[i] = r;
}

__global__ void k_scatter(const float* __restrict__ boxes, const int* __restrict__ rank,
                          int n, float* __restrict__ tb, int k) {
    int i = blockIdx.x * blockDim.x + threadIdx.x;
    if (i >= n) return;
    int r = rank[i];
    if (r < k) {
        tb[(size_t)r * 4 + 0] = boxes[(size_t)i * 4 + 0];
        tb[(size_t)r * 4 + 1] = boxes[(size_t)i * 4 + 1];
        tb[(size_t)r * 4 + 2] = boxes[(size_t)i * 4 + 2];
        tb[(size_t)r * 4 + 3] = boxes[(size_t)i * 4 + 3];
    }
}

// Greedy NMS, single workgroup, keep-flags in LDS
__global__ void k_nms(const float* __restrict__ tb, char* __restrict__ keep,
                      int k, float th) {
    __shared__ char kp[6000];
    for (int i = threadIdx.x; i < k; i += blockDim.x) kp[i] = 1;
    __syncthreads();
    for (int i = 0; i < k; ++i) {
        if (kp[i]) {
            float x1i = tb[i * 4], y1i = tb[i * 4 + 1];
            float x2i = tb[i * 4 + 2], y2i = tb[i * 4 + 3];
            float ai = (x2i - x1i + 1.0f) * (y2i - y1i + 1.0f);
            for (int j = i + 1 + (int)threadIdx.x; j < k; j += blockDim.x) {
                if (!kp[j]) continue;
                float xj1 = tb[j * 4], yj1 = tb[j * 4 + 1];
                float xj2 = tb[j * 4 + 2], yj2 = tb[j * 4 + 3];
                float xx1 = fmaxf(x1i, xj1), yy1 = fmaxf(y1i, yj1);
                float xx2 = fminf(x2i, xj2), yy2 = fminf(y2i, yj2);
                float iw = fmaxf(xx2 - xx1 + 1.0f, 0.0f);
                float ih = fmaxf(yy2 - yy1 + 1.0f, 0.0f);
                float inter = iw * ih;
                float aj = (xj2 - xj1 + 1.0f) * (yj2 - yj1 + 1.0f);
                if (inter / (ai + aj - inter) > th) kp[j] = 0;
            }
        }
        __syncthreads();
    }
    for (int i = threadIdx.x; i < k; i += blockDim.x) keep[i] = kp[i];
}

__global__ void k_select(const char* __restrict__ keep, const float* __restrict__ tb,
                         float* __restrict__ rois, int k, int post) {
    if (threadIdx.x != 0 || blockIdx.x != 0) return;
    int c = 0;
    for (int i = 0; i < k && c < post; ++i)
        if (keep[i]) {
            rois[c * 5 + 0] = 0.0f;
            rois[c * 5 + 1] = tb[i * 4 + 0]; rois[c * 5 + 2] = tb[i * 4 + 1];
            rois[c * 5 + 3] = tb[i * 4 + 2]; rois[c * 5 + 4] = tb[i * 4 + 3];
            ++c;
        }
    for (int i = 0; i < k && c < post; ++i)
        if (!keep[i]) {
            rois[c * 5 + 0] = 0.0f;
            rois[c * 5 + 1] = tb[i * 4 + 0]; rois[c * 5 + 2] = tb[i * 4 + 1];
            rois[c * 5 + 3] = tb[i * 4 + 2]; rois[c * 5 + 4] = tb[i * 4 + 3];
            ++c;
        }
}

// ROI max pool 7x7 over HWC feat; pooled f16 [roi][c*49 + ph*7 + pw]
__global__ void k_roipool(const h16* __restrict__ feat, const float* __restrict__ rois,
                          h16* __restrict__ pooled, int C, int Hf, int Wf) {
    int roi = blockIdx.x, c = blockIdx.y;
    int bin = threadIdx.x;
    if (bin >= 49) return;
    int ph = bin / 7, pw = bin % 7;
    const float* R = rois + (size_t)roi * 5;
    int x1 = (int)rintf(R[1] * 0.0625f);
    int y1 = (int)rintf(R[2] * 0.0625f);
    int x2 = (int)rintf(R[3] * 0.0625f);
    int y2 = (int)rintf(R[4] * 0.0625f);
    float rh = (float)imaxi(y2 - y1 + 1, 1);
    float rw = (float)imaxi(x2 - x1 + 1, 1);
    float bh = rh / 7.0f, bw = rw / 7.0f;
    int hs = imini(imaxi((int)floorf(ph * bh) + y1, 0), Hf);
    int he = imini(imaxi((int)ceilf((ph + 1) * bh) + y1, 0), Hf);
    int ws = imini(imaxi((int)floorf(pw * bw) + x1, 0), Wf);
    int we = imini(imaxi((int)ceilf((pw + 1) * bw) + x1, 0), Wf);
    float v = -__builtin_inff();
    for (int yy = hs; yy < he; ++yy)
        for (int xx = ws; xx < we; ++xx)
            v = fmaxf(v, (float)feat[((size_t)yy * Wf + xx) * C + c]);
    if (he <= hs || we <= ws) v = 0.0f;
    pooled[((size_t)roi * C + c) * 49 + ph * 7 + pw] = (h16)v;
}

// ---------------------------------------------------------------------------
// FC GEMM with LDS-staged weights (f32 -> f16 conversion in fill stage).
// A f16 [M][K], W f32 [N][K], Y f16 [M][N]. Block = 8 waves, one 64-col group.
// ---------------------------------------------------------------------------
__global__ __launch_bounds__(256)
void k_gemm_wmma_lds(const h16* __restrict__ A, const float* __restrict__ Wt,
                     const float* __restrict__ bias, h16* __restrict__ Y,
                     int M, int N, int K, int relu) {
    __shared__ h16 ldsB[2][64 * 32];
    const int tid = threadIdx.x;
    const int waveid = tid >> 5;
    const int lane = tid & 31;
    const int g = lane >> 4;
    const int l15 = lane & 15;
    const int mt = blockIdx.x * 8 + waveid;
    const int ngBase = blockIdx.y * 64;
    const int m = mt * 16 + l15;
    const bool okA = (m < M);
    const h16* pa = A + (size_t)(okA ? m : 0) * K + (g << 4);

    const int fch = tid >> 2;                    // 0..63
    const int fpart = (tid & 3) << 3;            // 0,8,16,24
    const float* wrow = Wt + (size_t)(ngBase + fch) * K + fpart;

    // fill chunk 0: 8 f32 -> 8 f16 -> ds_store_b128
    {
        f4 q0 = *(const f4*)(wrow);
        f4 q1 = *(const f4*)(wrow + 4);
        v8h hv;
#pragma unroll
        for (int e = 0; e < 4; ++e) { hv[e] = (h16)q0[e]; hv[4 + e] = (h16)q1[e]; }
        *(v8h*)&ldsB[0][fch * 32 + fpart] = hv;
    }
    __syncthreads();

    v8f acc0 = {}, acc1 = {}, acc2 = {}, acc3 = {};
    const v8h z8 = {};
    const int nchunk = K >> 5;

    for (int c = 0; c < nchunk; ++c) {
        const int kb = c << 5;
        if (c + 1 < nchunk) {
            if (kb + 1056 < K) __builtin_prefetch(wrow + kb + 1056, 0, 0); // global_prefetch_b8
            f4 q0 = *(const f4*)(wrow + kb + 32);
            f4 q1 = *(const f4*)(wrow + kb + 36);
            v8h hv;
#pragma unroll
            for (int e = 0; e < 4; ++e) { hv[e] = (h16)q0[e]; hv[4 + e] = (h16)q1[e]; }
            *(v8h*)&ldsB[(c + 1) & 1][fch * 32 + fpart] = hv;
        }
        U16h a;
        a.h[0] = z8;
        a.h[1] = z8;
        if (okA) {
            a.h[0] = *(const v8h*)(pa + kb);
            a.h[1] = *(const v8h*)(pa + kb + 8);
        }
        const h16* lb = &ldsB[c & 1][g << 4];
        U16h b0, b1, b2, b3;
        b0.h[0] = *(const v8h*)(lb + (l15 +  0) * 32);
        b0.h[1] = *(const v8h*)(lb + (l15 +  0) * 32 + 8);
        b1.h[0] = *(const v8h*)(lb + (l15 + 16) * 32);
        b1.h[1] = *(const v8h*)(lb + (l15 + 16) * 32 + 8);
        b2.h[0] = *(const v8h*)(lb + (l15 + 32) * 32);
        b2.h[1] = *(const v8h*)(lb + (l15 + 32) * 32 + 8);
        b3.h[0] = *(const v8h*)(lb + (l15 + 48) * 32);
        b3.h[1] = *(const v8h*)(lb + (l15 + 48) * 32 + 8);
        acc0 = __builtin_amdgcn_wmma_f32_16x16x32_f16(false, a.v, false, b0.v, (short)0, acc0, false, false);
        acc1 = __builtin_amdgcn_wmma_f32_16x16x32_f16(false, a.v, false, b1.v, (short)0, acc1, false, false);
        acc2 = __builtin_amdgcn_wmma_f32_16x16x32_f16(false, a.v, false, b2.v, (short)0, acc2, false, false);
        acc3 = __builtin_amdgcn_wmma_f32_16x16x32_f16(false, a.v, false, b3.v, (short)0, acc3, false, false);
        __syncthreads();
    }

    const int om0 = mt * 16 + (g << 3);
    v8f accs[4] = {acc0, acc1, acc2, acc3};
#pragma unroll
    for (int t = 0; t < 4; ++t) {
        const int n = ngBase + t * 16 + l15;
        const float bv = bias[n];
#pragma unroll
        for (int r = 0; r < 8; ++r) {
            int om = om0 + r;
            if (om < M) {
                float v = accs[t][r] + bv;
                if (relu) v = fmaxf(v, 0.0f);
                Y[(size_t)om * N + n] = (h16)v;
            }
        }
    }
}

// Tiny heads (N = 21 / 84): scalar, f16 A, f32 W/out
__global__ void k_fc_small(const h16* __restrict__ A, const float* __restrict__ Wt,
                           const float* __restrict__ b, float* __restrict__ Y,
                           int M, int N, int K) {
    int t = blockIdx.x * blockDim.x + threadIdx.x;
    if (t >= M * N) return;
    int n = t % N, m = t / N;
    const h16* ar = A + (size_t)m * K;
    const float* wr = Wt + (size_t)n * K;
    float s = b[n];
    for (int k = 0; k < K; ++k) s += (float)ar[k] * wr[k];
    Y[t] = s;
}

__global__ void k_softmax_rows(float* __restrict__ Y, int M, int N) {
    int m = blockIdx.x * blockDim.x + threadIdx.x;
    if (m >= M) return;
    float mx = -__builtin_inff();
    for (int i = 0; i < N; ++i) mx = fmaxf(mx, Y[(size_t)m * N + i]);
    float sm = 0.0f;
    for (int i = 0; i < N; ++i) {
        float e = expf(Y[(size_t)m * N + i] - mx);
        Y[(size_t)m * N + i] = e;
        sm += e;
    }
    float inv = 1.0f / sm;
    for (int i = 0; i < N; ++i) Y[(size_t)m * N + i] *= inv;
}

// ---------------------------------------------------------------------------
// Host side
// ---------------------------------------------------------------------------
static void gen_anchors(float out[36]) {
    const double ratios[3] = {0.5, 1.0, 2.0};
    const double scales[3] = {8.0, 16.0, 32.0};
    const double xc = 7.5, yc = 7.5, size = 256.0;
    int idx = 0;
    for (int r = 0; r < 3; ++r) {
        double ws = std::nearbyint(std::sqrt(size / ratios[r]));
        double hs = std::nearbyint(ws * ratios[r]);
        double rx1 = xc - 0.5 * (ws - 1), ry1 = yc - 0.5 * (hs - 1);
        double rx2 = xc + 0.5 * (ws - 1), ry2 = yc + 0.5 * (hs - 1);
        double w2 = rx2 - rx1 + 1, h2 = ry2 - ry1 + 1;
        double xc2 = rx1 + 0.5 * (w2 - 1), yc2 = ry1 + 0.5 * (h2 - 1);
        for (int s = 0; s < 3; ++s) {
            double ws2 = w2 * scales[s], hs2 = h2 * scales[s];
            out[idx * 4 + 0] = (float)(xc2 - 0.5 * (ws2 - 1));
            out[idx * 4 + 1] = (float)(yc2 - 0.5 * (hs2 - 1));
            out[idx * 4 + 2] = (float)(xc2 + 0.5 * (ws2 - 1));
            out[idx * 4 + 3] = (float)(yc2 + 0.5 * (hs2 - 1));
            ++idx;
        }
    }
}

static inline int cdiv(long long a, long long b) { return (int)((a + b - 1) / b); }

extern "C" void kernel_launch(void* const* d_in, const int* in_sizes, int n_in,
                              void* d_out, int out_size, void* d_ws, size_t ws_size,
                              hipStream_t stream) {
    (void)in_sizes; (void)n_in; (void)out_size; (void)ws_size;
    const float* im      = (const float*)d_in[0];
    const float* im_info = (const float*)d_in[1];
    const float* rpn_w   = (const float*)d_in[28];
    const float* rpn_b   = (const float*)d_in[29];
    const float* score_w = (const float*)d_in[30];
    const float* score_b = (const float*)d_in[31];
    const float* bbox_w  = (const float*)d_in[32];
    const float* bbox_b  = (const float*)d_in[33];
    const float* fc6_w   = (const float*)d_in[34];
    const float* fc6_b   = (const float*)d_in[35];
    const float* fc7_w   = (const float*)d_in[36];
    const float* fc7_b   = (const float*)d_in[37];
    const float* cls_w   = (const float*)d_in[38];
    const float* cls_b   = (const float*)d_in[39];
    const float* breg_w  = (const float*)d_in[40];
    const float* breg_b  = (const float*)d_in[41];

    char* ws = (char*)d_ws;
    size_t off = 0;
    auto carve = [&](size_t bytes) -> char* {
        off = (off + 255) & ~(size_t)255;
        char* p = ws + off;
        off += bytes;
        return p;
    };
    const size_t PING = (size_t)64 * 600 * 1000 * sizeof(h16);
    h16*   P0  = (h16*)carve(PING);
    h16*   P1  = (h16*)carve(PING);
    h16*   WPK = (h16*)carve((size_t)512 * 9 * 512 * sizeof(h16));
    h16*   RB  = (h16*)carve((size_t)512 * 37 * 62 * sizeof(h16));
    float* SC  = (float*)carve((size_t)18 * 37 * 62 * sizeof(float));
    float* BB  = (float*)carve((size_t)36 * 37 * 62 * sizeof(float));
    float* SCR = (float*)carve((size_t)37 * 62 * 9 * sizeof(float));
    float* BX  = (float*)carve((size_t)37 * 62 * 9 * 4 * sizeof(float));
    int*   RK  = (int*)carve((size_t)37 * 62 * 9 * sizeof(int));
    float* TB  = (float*)carve((size_t)6000 * 4 * sizeof(float));
    char*  KP  = (char*)carve(6000);
    h16*   PL  = (h16*)carve((size_t)300 * 512 * 49 * sizeof(h16));
    h16*   H6  = (h16*)carve((size_t)300 * 4096 * sizeof(h16));
    h16*   H7  = (h16*)carve((size_t)300 * 4096 * sizeof(h16));

    int H = 600, W = 1000;

    // input: NHWC f32 -> HWC f16 (identical element order)
    k_pack_input<<<cdiv(3LL * H * W, 256), 256, 0, stream>>>(im, P0, 3 * H * W);

    // VGG16 backbone
    const int cfg_out[13]    = {64, 64, 128, 128, 256, 256, 256, 512, 512, 512, 512, 512, 512};
    const int pool_after[13] = {0, 1, 0, 1, 0, 0, 1, 0, 0, 1, 0, 0, 0};
    h16* cur = P0;
    h16* nxt = P1;
    int cin = 3;
    for (int i = 0; i < 13; ++i) {
        const float* wv = (const float*)d_in[2 + 2 * i];
        const float* bv = (const float*)d_in[3 + 2 * i];
        int cout = cfg_out[i];
        int K = cin * 9;
        k_pack_w3x3<<<cdiv((long long)K * cout, 256), 256, 0, stream>>>(wv, WPK, cin, cout);
        int HW = H * W;
        int Mtiles = cdiv(HW, 16);
        if (cin % 32 == 0 && cout % 64 == 0) {
            dim3 grid(cdiv(Mtiles, 8), cout / 64);
            k_conv3x3_wmma_lds<<<grid, 256, 0, stream>>>(cur, WPK, bv, nxt, cin, H, W, cout, 1);
        } else {
            int waves = Mtiles * (cout / 16);
            k_conv3x3_any<<<cdiv(waves, 8), 256, 0, stream>>>(cur, WPK, bv, nxt, cin, H, W, cout, 1);
        }
        { h16* t = cur; cur = nxt; nxt = t; }
        cin = cout;
        if (pool_after[i]) {
            int Ho = H / 2, Wo = W / 2;
            long long total = (long long)cout * Ho * Wo;
            k_maxpool<<<cdiv(total, 256), 256, 0, stream>>>(cur, nxt, cout, H, W);
            { h16* t = cur; cur = nxt; nxt = t; }
            H = Ho; W = Wo;
        }
    }
    h16* feat = cur;                    // HWC, 37 x 62 x 512
    const int Hf = H, Wf = W, HWf = H * W;

    // RPN head
    k_pack_w3x3<<<cdiv((long long)512 * 9 * 512, 256), 256, 0, stream>>>(rpn_w, WPK, 512, 512);
    {
        dim3 grid(cdiv(cdiv(HWf, 16), 8), 512 / 64);
        k_conv3x3_wmma_lds<<<grid, 256, 0, stream>>>(feat, WPK, rpn_b, RB, 512, Hf, Wf, 512, 1);
    }
    k_conv1x1<<<cdiv(18LL * HWf, 256), 256, 0, stream>>>(RB, score_w, score_b, SC, 512, HWf, 18);
    k_conv1x1<<<cdiv(36LL * HWf, 256), 256, 0, stream>>>(RB, bbox_w, bbox_b, BB, 512, HWf, 36);

    // proposals
    Anchors AN;
    gen_anchors(AN.a);
    int n_anchors = HWf * 9;
    k_decode<<<cdiv(n_anchors, 256), 256, 0, stream>>>(SC, BB, im_info, AN, SCR, BX, Hf, Wf);
    k_rank<<<cdiv(n_anchors, 256), 256, 0, stream>>>(SCR, n_anchors, RK);
    k_scatter<<<cdiv(n_anchors, 256), 256, 0, stream>>>(BX, RK, n_anchors, TB, 6000);
    k_nms<<<1, 1024, 0, stream>>>(TB, KP, 6000, 0.7f);

    float* out_cls  = (float*)d_out;             // 300 x 21
    float* out_breg = out_cls + 300 * 21;        // 300 x 84
    float* out_rois = out_breg + 300 * 84;       // 300 x 5
    k_select<<<1, 1, 0, stream>>>(KP, TB, out_rois, 6000, 300);

    // ROI pool + FC head
    {
        dim3 g(300, 512);
        k_roipool<<<g, 64, 0, stream>>>(feat, out_rois, PL, 512, Hf, Wf);
    }
    {
        dim3 grid6(cdiv(cdiv(300, 16), 8), 4096 / 64);
        k_gemm_wmma_lds<<<grid6, 256, 0, stream>>>(PL, fc6_w, fc6_b, H6, 300, 4096, 512 * 49, 1);
        k_gemm_wmma_lds<<<grid6, 256, 0, stream>>>(H6, fc7_w, fc7_b, H7, 300, 4096, 4096, 1);
    }
    k_fc_small<<<cdiv(300 * 21, 256), 256, 0, stream>>>(H7, cls_w, cls_b, out_cls, 300, 21, 4096);
    k_softmax_rows<<<cdiv(300, 64), 64, 0, stream>>>(out_cls, 300, 21);
    k_fc_small<<<cdiv(300 * 84, 256), 256, 0, stream>>>(H7, breg_w, breg_b, out_breg, 300, 84, 4096);
}
